// GraphAttention_75067438399950
// MI455X (gfx1250) — matrix operand order repaired and checked by reading, not compiled, not run on hardware
//
#include <hip/hip_runtime.h>
#include <hip/hip_bf16.h>

typedef __attribute__((ext_vector_type(16))) __bf16 v16bf;
typedef __attribute__((ext_vector_type(8)))  __bf16 v8bf;
typedef __attribute__((ext_vector_type(8)))  float  v8f;
typedef __attribute__((ext_vector_type(4)))  unsigned int u32x4;
typedef __attribute__((ext_vector_type(8)))  unsigned int u32x8;

#define NNODES 4096
#define INF    128
#define NHID   64
#define NBATCH 16
#define NCOL   (NBATCH * NHID)   // 1024 folded columns (b,d)

// -------- Kernel 1: h = xW + b  (bf16 WMMA, f32 accum) -----------------------
// Writes h (f32, row-major (b,n,d)) and H (bf16, node-major: H[n][b*64+d])
__global__ void __launch_bounds__(256)
k_hidden(const float* __restrict__ x, const float* __restrict__ W,
         const float* __restrict__ bias, float* __restrict__ h,
         __bf16* __restrict__ H) {
  const int tid  = threadIdx.x;
  const int wave = tid >> 5;
  const int lane = tid & 31;
  const int m    = lane & 15;
  const int half = lane >> 4;

  const int row_tile = blockIdx.x * 32 + (wave >> 2) * 16;  // 2 row tiles
  const int col_tile = (wave & 3) * 16;                     // 4 col tiles

  const int r = row_tile + m;                   // A row this lane supplies
  // reference zeroes x[:,128,:]; fold into a 0/1 scale so loads stay b128
  const float zs = ((r & (NNODES - 1)) == INF) ? 0.0f : 1.0f;

  v8f acc = {};
#pragma unroll
  for (int k0 = 0; k0 < INF; k0 += 32) {
    float fa[16], fb[16];
    const float4* ap = (const float4*)(x + (size_t)r * INF + k0 + half * 8);
    *(float4*)&fa[0]  = ap[0];   // K = half*8 + 0..7
    *(float4*)&fa[4]  = ap[1];
    *(float4*)&fa[8]  = ap[4];   // K = 16 + half*8 + 0..7
    *(float4*)&fa[12] = ap[5];
#pragma unroll
    for (int e = 0; e < 16; ++e)   // B col N = m, K in same order as A
      fb[e] = W[(size_t)(k0 + (e >> 3) * 16 + half * 8 + (e & 7)) * NHID + col_tile + m];
    v16bf a, bm;
#pragma unroll
    for (int e = 0; e < 16; ++e) {
      a[e]  = (__bf16)(fa[e] * zs);
      bm[e] = (__bf16)fb[e];
    }
    acc = __builtin_amdgcn_wmma_f32_16x16x32_bf16(false, a, false, bm,
                                                  (short)0, acc, false, false);
  }
  const int col = col_tile + m;     // C: N = lane%16
  const float bv = bias[col];
#pragma unroll
  for (int v = 0; v < 8; ++v) {
    const int row = row_tile + v + 8 * half;   // C: M = v + 8*half
    const float val = acc[v] + bv;
    h[(size_t)row * NHID + col] = val;
    const int bb = row >> 12, nn = row & (NNODES - 1);
    H[(size_t)nn * NCOL + bb * NHID + col] = (__bf16)val;
  }
}

// -------- Kernel 2: agg = maskT(4096x4096) x H(4096x1024)  ------------------
// One GEMM for all 16 batches (mask is batch-independent). mask row 4095 == 0
// implements the reference's h[:, :-1] slice. Double-buffered software
// pipeline: TDM (tensor_load_to_lds) fetches the next H tile and the next
// mask tile is converted int32->bf16 while the current tiles feed 8 WMMAs
// per wave; one barrier per stage.
#define MBLK   64
#define NBLK   128
#define KSTAGE 64
#define NSTAGE (NNODES / KSTAGE)   // 64

// TDM D#: 2D bf16 tile, NBLK cols x KSTAGE rows, tensor row stride NCOL;
// LDS pad 4 dwords after every 64 dwords -> reproduces [KSTAGE][NBLK+8].
__device__ __forceinline__ void tdm_issue(const __bf16* src, const __bf16* dst) {
  unsigned long long ga = (unsigned long long)(size_t)src;
  u32x4 g0;
  g0[0] = 1u;                                    // count=1, user descriptor
  g0[1] = (unsigned int)(size_t)dst;             // lds_addr
  g0[2] = (unsigned int)ga;                      // global_addr[31:0]
  g0[3] = (unsigned int)((ga >> 32) & 0x1FFFFFFull) | (2u << 30);  // type=2
  u32x8 g1;
  g1[0] = (1u << 16)          // data_size = 2 bytes
        | (1u << 20)          // pad_enable
        | (5u << 22)          // pad_interval: 64 dwords
        | (3u << 25);         // pad_amount: 4 dwords (= 8 bf16)
  g1[1] = (NCOL & 0xFFFFu) << 16;                    // tensor_dim0 lo
  g1[2] = (NCOL >> 16) | ((NNODES & 0xFFFFu) << 16); // dim0 hi | dim1 lo
  g1[3] = (NNODES >> 16) | ((unsigned)NBLK << 16);   // dim1 hi | tile_dim0
  g1[4] = KSTAGE;                                    // tile_dim1 | tile_dim2=0
  g1[5] = NCOL;                                      // tensor_dim0_stride lo
  g1[6] = 0;
  g1[7] = 0;
  asm volatile("tensor_load_to_lds %0, %1" :: "s"(g0), "s"(g1) : "memory");
}

// Stage transposed mask tile: At[i][j] = adj[j+1][i_blk+i] > 0 (bf16, exact),
// plus cache prefetch of the stage after it.
__device__ __forceinline__ void stage_mask(const int* __restrict__ adj,
                                           int i_blk, int st, int tid,
                                           __bf16 (*At)[KSTAGE + 16]) {
  const int k0 = st * KSTAGE;
  for (int idx = tid; idx < KSTAGE * MBLK; idx += 256) {
    const int jl = idx >> 6, il = idx & 63;   // coalesced adj reads over il
    const int j = k0 + jl;
    int v = 0;
    if (j < NNODES - 1) v = adj[(size_t)(j + 1) * NNODES + i_blk + il] > 0;
    At[il][jl] = (__bf16)(float)v;
  }
  if (st + 1 < NSTAGE) {
    const int j = k0 + KSTAGE + (tid >> 3);
    __builtin_prefetch(&adj[(size_t)(j + 1) * NNODES + i_blk + (tid & 7) * 8], 0, 0);
  }
}

__global__ void __launch_bounds__(256)
k_agg(const int* __restrict__ adj, const __bf16* __restrict__ H,
      float* __restrict__ agg) {
  __shared__ alignas(16) __bf16 At[2][MBLK][KSTAGE + 16];  // [i][k], 160B rows
  __shared__ alignas(16) __bf16 Hs[2][KSTAGE][NBLK + 8];   // [k][bd], 272B rows

  const int tid  = threadIdx.x;
  const int wave = tid >> 5;
  const int lane = tid & 31;
  const int m    = lane & 15;
  const int half = lane >> 4;

  const int i_blk = blockIdx.x * MBLK;
  const int n_blk = blockIdx.y * NBLK;
  const int mw = (wave >> 2) * 32;   // wave row offset: 0 or 32
  const int nw = (wave & 3) * 32;    // wave col offset: 0..96

  v8f acc[2][2] = {};

  // ---- pipeline prologue: fill buffer 0 ----
  if (wave == 0) tdm_issue(H + n_blk, &Hs[0][0][0]);
  stage_mask(adj, i_blk, 0, tid, At[0]);
  if (wave == 0) __builtin_amdgcn_s_wait_tensorcnt(0);
  __syncthreads();

  for (int s = 0; s < NSTAGE; ++s) {
    const int cur = s & 1;
    // ---- kick off stage s+1 into the other buffer ----
    if (s + 1 < NSTAGE) {
      if (wave == 0)
        tdm_issue(H + (size_t)(s + 1) * KSTAGE * NCOL + n_blk,
                  &Hs[cur ^ 1][0][0]);
      stage_mask(adj, i_blk, s + 1, tid, At[cur ^ 1]);
    }

    // ---- compute stage s from buffer cur ----
#pragma unroll
    for (int ks = 0; ks < 2; ++ks) {
      v16bf afrag[2], bfrag[2];
#pragma unroll
      for (int sm = 0; sm < 2; ++sm) {
        const int mo = mw + sm * 16 + m;
        const v8bf* lo = (const v8bf*)&At[cur][mo][ks * 32 + half * 8];
        const v8bf* hi = (const v8bf*)&At[cur][mo][ks * 32 + 16 + half * 8];
        afrag[sm] = __builtin_shufflevector(
            lo[0], hi[0], 0, 1, 2, 3, 4, 5, 6, 7, 8, 9, 10, 11, 12, 13, 14, 15);
      }
#pragma unroll
      for (int sn = 0; sn < 2; ++sn) {
#pragma unroll
        for (int e = 0; e < 16; ++e)
          bfrag[sn][e] = Hs[cur][ks * 32 + (e >> 3) * 16 + half * 8 + (e & 7)][nw + sn * 16 + m];
      }
#pragma unroll
      for (int sm = 0; sm < 2; ++sm)
#pragma unroll
        for (int sn = 0; sn < 2; ++sn)
          acc[sm][sn] = __builtin_amdgcn_wmma_f32_16x16x32_bf16(
              false, afrag[sm], false, bfrag[sn], (short)0, acc[sm][sn],
              false, false);
    }

    // drain the async tile before anyone can read it next stage, then the
    // single per-stage barrier hands both buffers over.
    if (wave == 0 && s + 1 < NSTAGE) __builtin_amdgcn_s_wait_tensorcnt(0);
    __syncthreads();
  }

#pragma unroll
  for (int sm = 0; sm < 2; ++sm)
#pragma unroll
    for (int sn = 0; sn < 2; ++sn) {
      const int colg = n_blk + nw + sn * 16 + m;   // folded (b,d) column
      const int bb = colg >> 6, d = colg & 63;
#pragma unroll
      for (int v = 0; v < 8; ++v) {
        const int i = i_blk + mw + sm * 16 + v + 8 * half;
        agg[((size_t)bb * NNODES + i) * NHID + d] = acc[sm][sn][v];
      }
    }
}

// -------- Kernel 3: h_sum[b][d] = sum_n h[b][n][d] --------------------------
__global__ void __launch_bounds__(256)
k_hsum(const float* __restrict__ h, float* __restrict__ hsum) {
  __shared__ float part[4][NHID];
  const int b = blockIdx.x;
  const int d = threadIdx.x & 63;
  const int p = threadIdx.x >> 6;
  float s = 0.f;
  for (int n = p; n < NNODES; n += 4)
    s += h[((size_t)b * NNODES + n) * NHID + d];
  part[p][d] = s;
  __syncthreads();
  if (threadIdx.x < NHID)
    hsum[b * NHID + d] = part[0][d] + part[1][d] + part[2][d] + part[3][d];
}

// -------- Kernel 4: score, outer product, leaky ReLU ------------------------
__global__ void __launch_bounds__(256)
k_out(const float* __restrict__ h, const float* __restrict__ agg,
      const float* __restrict__ a, const float* __restrict__ hsum,
      float* __restrict__ out) {
  const int wave = threadIdx.x >> 5;
  const int lane = threadIdx.x & 31;
  const int r = blockIdx.x * 8 + wave;        // flat (b,i)
  const int b = r >> 12;
  const int i = r & (NNODES - 1);
  const int d0 = lane * 2;

  float s = 0.f;
#pragma unroll
  for (int q = 0; q < 2; ++q) {
    const int d = d0 + q;
    s += h  [(size_t)r * NHID + d] * a[(size_t)d          * NNODES + i]
       + agg[(size_t)r * NHID + d] * a[(size_t)(NHID + d) * NNODES + i];
  }
#pragma unroll
  for (int off = 16; off >= 1; off >>= 1)   // wave32 reduction
    s += __shfl_xor(s, off, 32);
#pragma unroll
  for (int q = 0; q < 2; ++q) {
    const int d = d0 + q;
    const float v = s * hsum[b * NHID + d];
    out[(size_t)r * NHID + d] = v >= 0.f ? v : 0.01f * v;
  }
}

extern "C" void kernel_launch(void* const* d_in, const int* in_sizes, int n_in,
                              void* d_out, int out_size, void* d_ws, size_t ws_size,
                              hipStream_t stream) {
  const float* x    = (const float*)d_in[0];
  const int*   adj  = (const int*)  d_in[1];
  const float* W    = (const float*)d_in[2];
  const float* bias = (const float*)d_in[3];
  const float* a    = (const float*)d_in[4];

  char* ws = (char*)d_ws;
  float*  h    = (float*) ws;                          // 16 MB  f32 hidden
  __bf16* H    = (__bf16*)(ws + ((size_t)16 << 20));   //  8 MB  bf16 node-major
  float*  agg  = (float*) (ws + ((size_t)24 << 20));   // 16 MB  aggregation
  float*  hsum = (float*) (ws + ((size_t)40 << 20));   //  4 KB

  k_hidden<<<(NBATCH * NNODES) / 32, 256, 0, stream>>>(x, W, bias, h, H);

  dim3 g2(NNODES / MBLK, NCOL / NBLK);                 // 64 x 8 blocks
  k_agg<<<g2, 256, 0, stream>>>(adj, H, agg);

  k_hsum<<<NBATCH, 256, 0, stream>>>(h, hsum);

  k_out<<<(NBATCH * NNODES) / 8, 256, 0, stream>>>(h, agg, a, hsum,
                                                   (float*)d_out);
}
